// VectorQuantizer_25262997635699
// MI455X (gfx1250) — compile-verified, matrix operand order checked by reading, not run on hardware
//
#include <hip/hip_runtime.h>
#include <hip/hip_bf16.h>

typedef __attribute__((ext_vector_type(16))) _Float16 v16h;
typedef __attribute__((ext_vector_type(2)))  __fp16   v2fp;
typedef __attribute__((ext_vector_type(8)))  float    v8f;
typedef int v4i_vec __attribute__((__vector_size__(16)));

#define VQ_D 64
#define VQ_K 1024
#define VQ_N (64 * 64 * 64)                 // 262144 rows
#define VQ_NZQ (64 * VQ_D * 64 * 64)        // 16777216 floats of z_q
#define HW 4096                             // stride between d-planes
#define ROWS_PER_BLOCK 128                  // 4 waves x 32 rows
#define NTHREADS 128
#define ZSTR 132                            // padded f32 LDS stride per d-plane
#define GROUP 8                             // code tiles per staging group
#define NGROUPS 8                           // 64 tiles total
#define GROUP_HALVES (GROUP * 1024)         // 8192 halves = 16 KB per group

#if defined(__has_builtin)
#if __has_builtin(__builtin_amdgcn_global_load_async_to_lds_b128)
#define VQ_HAVE_ASYNC 1
#endif
#endif
#ifndef VQ_HAVE_ASYNC
#define VQ_HAVE_ASYNC 0
#endif

#if VQ_HAVE_ASYNC
#if __has_builtin(__builtin_amdgcn_s_wait_asynccnt)
#define VQ_WAIT_ASYNC(n) __builtin_amdgcn_s_wait_asynccnt(n)
#else
#define VQ_WAIT_ASYNC(n) asm volatile("s_wait_asynccnt %0" ::"i"(n) : "memory")
#endif
#define VQ_ASYNC_B128(g, l)                                                    \
    __builtin_amdgcn_global_load_async_to_lds_b128(                            \
        (__attribute__((address_space(1))) v4i_vec*)(g),                       \
        (__attribute__((address_space(3))) v4i_vec*)(l), 0, 0)
#else
#define VQ_WAIT_ASYNC(n)
#endif

// K-local index inside a 32-wide chunk for half-pair j (A & B share this
// swizzle per the CDNA5 ISA 16-bit matrix VGPR layout):
//   j<4 : K = 2j + 8*hi  (VGPR0-3) ; j>=4: K = 16+2(j-4)+8*hi (VGPR4-7)
__device__ __forceinline__ int klocal(int j, int hi) {
    return (j < 4) ? (2 * j + 8 * hi) : (16 + 2 * (j - 4) + 8 * hi);
}

__device__ __forceinline__ unsigned pack2h(float a, float b) {
#if defined(__has_builtin) && __has_builtin(__builtin_amdgcn_cvt_pkrtz)
    union { v2fp v; unsigned u; } c;
    c.v = __builtin_amdgcn_cvt_pkrtz(a, b);
    return c.u;
#else
    _Float16 lo = (_Float16)a, hi = (_Float16)b;
    unsigned ul = *(unsigned short*)&lo, uh = *(unsigned short*)&hi;
    return ul | (uh << 16);
#endif
}

// ---------------------------------------------------------------------------
// Prep: codebook f32 [K, D] -> pre-swizzled f16 B fragments + (-0.5*||c||^2).
// frag layout: ((tile*2 + chunk)*32 + lane) * 16 halves (32B/lane, coalesced)
// ---------------------------------------------------------------------------
__global__ void vq_prep_codebook(const float* __restrict__ cb,
                                 _Float16* __restrict__ frag,
                                 float* __restrict__ cbhn) {
    const int t    = blockIdx.x;   // code tile (16 codes)
    const int lane = threadIdx.x;  // 32 lanes
    const int n    = lane & 15;
    const int hi   = lane >> 4;
    const int code = t * 16 + n;

    #pragma unroll
    for (int c = 0; c < 2; ++c) {
        union { unsigned u[8]; v16h v; } pk;
        #pragma unroll
        for (int j = 0; j < 8; ++j) {
            int d = c * 32 + klocal(j, hi);
            pk.u[j] = pack2h(cb[code * VQ_D + d], cb[code * VQ_D + d + 1]);
        }
        *(v16h*)(frag + ((size_t)(t * 2 + c) * 32 + lane) * 16) = pk.v;
    }

    if (lane < 16) {
        float s = 0.f;
        #pragma unroll 8
        for (int d = 0; d < VQ_D; ++d) {
            float v = cb[code * VQ_D + d];
            s += v * v;
        }
        cbhn[code] = -0.5f * s;  // pre-negated half-norm: WMMA C seed
    }
}

// ---------------------------------------------------------------------------
// Main: 4 waves x 32 rows per block vs all 1024 codes. B fragments are
// streamed through LDS (async, double-buffered 16KB groups) so each block
// pulls the 128KB fragment set from L2 exactly once; each wave amortizes
// every B fragment over two 16-row A tiles (4 WMMAs per code tile).
// ---------------------------------------------------------------------------
union VqShMem {
    float    zf[VQ_D * ZSTR];            // 33 KB f32 z tile, [d][row] (phase 1)
    _Float16 bbuf[2][GROUP_HALVES];      // 32 KB B double buffer   (phase 2)
};

__global__ void __launch_bounds__(NTHREADS)
vq_main(const float* __restrict__ z,       // [B, D, H, W]
        const float* __restrict__ cb,      // [K, D] f32 (for the gather)
        const _Float16* __restrict__ frag, // swizzled B fragments
        const float* __restrict__ cbhn,    // [K] = -0.5*||c||^2
        float* __restrict__ zq_out,        // [B, D, H, W]
        int* __restrict__ idx_out) {       // [N]
    __shared__ VqShMem sh;
    __shared__ float lds_hn[VQ_K];            // 4 KB seeds
    __shared__ int   lds_idx[ROWS_PER_BLOCK]; // winners

    const int tid  = threadIdx.x;
    const int lane = tid & 31;
    const int wave = tid >> 5;
    const int m    = lane & 15;
    const int hi   = lane >> 4;

    const int    n0 = blockIdx.x * ROWS_PER_BLOCK;
    const int    bi = n0 >> 12;       // batch index (4096 rows per batch)
    const int    s0 = n0 & (HW - 1);  // spatial offset inside batch
    const float* zb = z + (size_t)bi * (VQ_D * HW) + s0;

    // Stage WMMA-C seeds (coalesced).
    for (int i = tid; i < VQ_K; i += NTHREADS) lds_hn[i] = cbhn[i];

    // ---- Phase 1: stage 128x64 z tile as f32, layout [d][row]. One async
    // B128 per lane-quad moves a whole 512B d-plane.
#if VQ_HAVE_ASYNC
    {
        const int dbase = wave * 16;  // 16 d-planes per wave
        #pragma unroll
        for (int i = 0; i < 16; ++i) {
            int d = dbase + i;
            VQ_ASYNC_B128(zb + (size_t)d * HW + lane * 4,
                          &sh.zf[d * ZSTR + lane * 4]);
        }
    }
    VQ_WAIT_ASYNC(0);
    __syncthreads();
#else
    for (int it = 0; it < 64; ++it) {
        int idx = it * NTHREADS + tid;
        int d = idx >> 7, col = idx & 127;
        sh.zf[d * ZSTR + col] =
            __builtin_nontemporal_load(&zb[(size_t)d * HW + col]);
    }
    __syncthreads();
#endif

    // Build four persistent A fragments: two 16-row M tiles x two 32-d chunks,
    // packed f32->f16. Bank-conflict-free (ZSTR % 64 == 4).
    v16h afrag[2][2];  // [mtile][chunk]
    #pragma unroll
    for (int mt = 0; mt < 2; ++mt) {
        const int rowcol = wave * 32 + mt * 16 + m;
        #pragma unroll
        for (int c = 0; c < 2; ++c) {
            union { unsigned u[8]; v16h v; } pk;
            #pragma unroll
            for (int j = 0; j < 8; ++j) {
                int d = c * 32 + klocal(j, hi);
                pk.u[j] = pack2h(sh.zf[d * ZSTR + rowcol],
                                 sh.zf[(d + 1) * ZSTR + rowcol]);
            }
            afrag[mt][c] = pk.v;
        }
    }
    __syncthreads();  // z tile dead; sh is reused as the B double buffer

    // ---- Phase 2: sweep all 64 code tiles, B staged via async double buffer.
    float bestv[2][8];
    int   besti[2][8];
    #pragma unroll
    for (int mt = 0; mt < 2; ++mt)
        #pragma unroll
        for (int r = 0; r < 8; ++r) { bestv[mt][r] = -3.4e38f; besti[mt][r] = 0; }

    // Each wave copies its 4KB quarter of a 16KB group (8 x 512B async B128).
    const char* fragb = (const char*)frag;
#if VQ_HAVE_ASYNC
    #pragma unroll
    for (int i = 0; i < 8; ++i)
        VQ_ASYNC_B128(fragb + wave * 4096 + i * 512 + lane * 16,
                      (char*)sh.bbuf[0] + wave * 4096 + i * 512 + lane * 16);
#else
    for (int i = 0; i < 8; ++i) {
        size_t o = (size_t)wave * 4096 + i * 512 + lane * 16;
        *(v4i_vec*)((char*)sh.bbuf[0] + o) = *(const v4i_vec*)(fragb + o);
    }
#endif

    for (int g = 0; g < NGROUPS; ++g) {
        const int buf = g & 1;
        if (g + 1 < NGROUPS) {
#if VQ_HAVE_ASYNC
            const char* gs = fragb + (size_t)(g + 1) * (GROUP_HALVES * 2);
            #pragma unroll
            for (int i = 0; i < 8; ++i)
                VQ_ASYNC_B128(gs + wave * 4096 + i * 512 + lane * 16,
                              (char*)sh.bbuf[buf ^ 1] + wave * 4096 + i * 512 +
                                  lane * 16);
            VQ_WAIT_ASYNC(8);  // in-order: group g's 8 copies have landed
#else
            const char* gs = fragb + (size_t)(g + 1) * (GROUP_HALVES * 2);
            for (int i = 0; i < 8; ++i) {
                size_t o = (size_t)wave * 4096 + i * 512 + lane * 16;
                *(v4i_vec*)((char*)sh.bbuf[buf ^ 1] + o) =
                    *(const v4i_vec*)(gs + o);
            }
#endif
        } else {
            VQ_WAIT_ASYNC(0);
        }
        __syncthreads();  // buffer[buf] fully populated by all waves

        const _Float16* bb = sh.bbuf[buf];
        #pragma unroll 2
        for (int i = 0; i < GROUP; ++i) {
            const int t = g * GROUP + i;
            v16h b0 = *(const v16h*)(bb + i * 1024 + lane * 16);
            v16h b1 = *(const v16h*)(bb + i * 1024 + 512 + lane * 16);
            float hn = lds_hn[t * 16 + m];

            v8f acc0, acc1;
            #pragma unroll
            for (int r = 0; r < 8; ++r) { acc0[r] = hn; acc1[r] = hn; }

            acc0 = __builtin_amdgcn_wmma_f32_16x16x32_f16(
                false, afrag[0][0], false, b0, (short)0, acc0, false, false);
            acc1 = __builtin_amdgcn_wmma_f32_16x16x32_f16(
                false, afrag[1][0], false, b0, (short)0, acc1, false, false);
            acc0 = __builtin_amdgcn_wmma_f32_16x16x32_f16(
                false, afrag[0][1], false, b1, (short)0, acc0, false, false);
            acc1 = __builtin_amdgcn_wmma_f32_16x16x32_f16(
                false, afrag[1][1], false, b1, (short)0, acc1, false, false);

            int cidx = t * 16 + m;
            #pragma unroll
            for (int r = 0; r < 8; ++r) {
                bool t0 = acc0[r] > bestv[0][r];
                bestv[0][r] = t0 ? acc0[r] : bestv[0][r];
                besti[0][r] = t0 ? cidx : besti[0][r];
                bool t1 = acc1[r] > bestv[1][r];
                bestv[1][r] = t1 ? acc1[r] : bestv[1][r];
                besti[1][r] = t1 ? cidx : besti[1][r];
            }
        }
        __syncthreads();  // all waves done reading buffer[buf]
    }

    // Branchless argmax reduction over the 16 code columns per 16-lane half;
    // ties resolve to the lower index (matches jnp.argmin's first hit).
    #pragma unroll
    for (int mt = 0; mt < 2; ++mt) {
        #pragma unroll
        for (int off = 8; off >= 1; off >>= 1) {
            #pragma unroll
            for (int r = 0; r < 8; ++r) {
                float ob = __shfl_xor(bestv[mt][r], off, 16);
                int   oi = __shfl_xor(besti[mt][r], off, 16);
                bool take = (ob > bestv[mt][r]) |
                            ((ob == bestv[mt][r]) & (oi < besti[mt][r]));
                bestv[mt][r] = take ? ob : bestv[mt][r];
                besti[mt][r] = take ? oi : besti[mt][r];
            }
        }
        if (m == 0) {  // lane 0 -> rows 0..7, lane 16 -> rows 8..15
            int base = wave * 32 + mt * 16 + hi * 8;
            #pragma unroll
            for (int r = 0; r < 8; ++r) lds_idx[base + r] = besti[mt][r];
        }
    }
    __syncthreads();

    // Emit indices (int32) and gather z_q back in [B, D, H, W] layout with
    // coalesced non-temporal stores; codebook rows are L2-hot.
    idx_out[n0 + tid] = lds_idx[tid];

    float* zqb  = zq_out + (size_t)bi * (VQ_D * HW) + s0;
    int    code = lds_idx[tid];
    #pragma unroll 4
    for (int d = 0; d < VQ_D; ++d) {
        float v = cb[(size_t)code * VQ_D + d];
        __builtin_nontemporal_store(v, &zqb[(size_t)d * HW + tid]);
    }
}

// ---------------------------------------------------------------------------
extern "C" void kernel_launch(void* const* d_in, const int* in_sizes, int n_in,
                              void* d_out, int out_size, void* d_ws, size_t ws_size,
                              hipStream_t stream) {
    (void)in_sizes; (void)n_in; (void)out_size; (void)ws_size;
    const float* z_e      = (const float*)d_in[0]; // [64,64,64,64]
    const float* codebook = (const float*)d_in[1]; // [1024,64]

    // Workspace: 128 KB swizzled f16 codebook fragments + 4 KB seeds.
    _Float16* frag = (_Float16*)d_ws;
    float*    cbhn = (float*)((char*)d_ws + (size_t)VQ_K * VQ_D * sizeof(_Float16));

    float* zq_out  = (float*)d_out;
    int*   idx_out = (int*)d_out + (size_t)VQ_NZQ;

    vq_prep_codebook<<<VQ_K / 16, 32, 0, stream>>>(codebook, frag, cbhn);
    vq_main<<<VQ_N / ROWS_PER_BLOCK, NTHREADS, 0, stream>>>(
        z_e, codebook, frag, cbhn, zq_out, idx_out);
}
